// RGAT_46239617908906
// MI455X (gfx1250) — matrix operand order repaired and checked by reading, not compiled, not run on hardware
//
#include <hip/hip_runtime.h>

// ---------------------------------------------------------------------------
// EGATConv x3 for MI455X (gfx1250): bf16 WMMA GEMMs (pre-packed B operand,
// zero-LDS GEMM) + fused edge kernels.
// ---------------------------------------------------------------------------

#define N_NODES 16000
#define N_EDGES 64000
#define DMODEL  256
#define NHEADS  8
#define SLOPE   0.01f

typedef __attribute__((ext_vector_type(16))) __bf16 v16bf;
typedef __attribute__((ext_vector_type(8)))  float  v8f;

// ----------------------------- helpers -------------------------------------

__device__ __forceinline__ __bf16 to_bf16(float f) {
    // round-to-nearest-even f32 -> bf16
    unsigned int u = __float_as_uint(f);
    unsigned int r = (u + 0x7FFFu + ((u >> 16) & 1u)) >> 16;
    unsigned short s = (unsigned short)r;
    return *reinterpret_cast<__bf16*>(&s);
}

__device__ __forceinline__ void atomicMaxFloat(float* addr, float value) {
    int* iaddr = (int*)addr;
    int old = *iaddr;
    while (true) {
        float fold = __int_as_float(old);
        if (fold >= value) break;
        int assumed = old;
        old = atomicCAS(iaddr, assumed, __float_as_int(value));
        if (old == assumed) break;
    }
}

// ------------------------ elementwise kernels -------------------------------

__global__ void f2bf_kernel(const float* __restrict__ in,
                            __bf16* __restrict__ out, size_t n) {
    size_t i = (size_t)blockIdx.x * blockDim.x + threadIdx.x;
    if (i < n) out[i] = to_bf16(in[i]);
}

// Convert a 256x256 column chunk of a [256, stride] f32 weight to bf16 AND
// pack it into the WMMA B-fragment-ready layout:
//   For k-tile kt (32 rows) and n-tile nt (16 cols), lane l holds 16 halfs
//   contiguous at ((kt*16 + nt)*32 + l)*16, with element j being
//   B[kt*32 + (j>>3)*16 + (l>>4)*8 + (j&7), nt*16 + (l&15)].
__global__ void wconv_pack_kernel(const float* __restrict__ W,
                                  __bf16* __restrict__ out,
                                  int stride, int coloff) {
    int i = blockIdx.x * blockDim.x + threadIdx.x; // 0..65535
    int j    = i & 15;
    int lane = (i >> 4) & 31;
    int nt   = (i >> 9) & 15;
    int kt   = i >> 13;                            // 0..7
    int k = kt * 32 + (j >> 3) * 16 + (lane >> 4) * 8 + (j & 7);
    int n = coloff + nt * 16 + (lane & 15);
    out[i] = to_bf16(W[(size_t)k * stride + n]);
}

__global__ void zero_kernel(float* __restrict__ p, size_t n) {
    size_t i = (size_t)blockIdx.x * blockDim.x + threadIdx.x;
    if (i < n) p[i] = 0.f;
}

__global__ void init_seg_kernel(float* __restrict__ segmax,
                                float* __restrict__ segsum, int n) {
    int i = blockIdx.x * blockDim.x + threadIdx.x;
    if (i < n) { segmax[i] = -3.0e38f; segsum[i] = 0.f; }
}

__global__ void node_update_kernel(float* __restrict__ n_cur,
                                   const float* __restrict__ acc,
                                   size_t n, int useElu) {
    size_t i = (size_t)blockIdx.x * blockDim.x + threadIdx.x;
    if (i < n) {
        float v = acc[i];
        n_cur[i] += useElu ? (v > 0.f ? v : __expf(v) - 1.f) : v;
    }
}

// ----------------------------- WMMA GEMM ------------------------------------
// C[M,256] = A[M,256](bf16, row-major) x B(bf16, fragment-packed), f32 acc.
// Block = 128 threads = 4 waves; each wave owns a 32x64 tile -> block tile
// 128x64. K-loop 8 x 32. No LDS: A frags are two contiguous 16B runs per
// lane, B frags are one contiguous 32B run per lane (pre-packed).

__global__ __launch_bounds__(128) void gemm_bf16_256(
    const __bf16* __restrict__ A, const __bf16* __restrict__ Bp,
    float* __restrict__ C, int M) {
    const int tid  = threadIdx.x;
    const int wave = tid >> 5;       // 0..3 (wave32)
    const int lane = tid & 31;
    const int m0   = blockIdx.x * 128 + wave * 32;
    const int nt0  = blockIdx.y * 4;             // first of 4 n-tiles (of 16)
    const int mrow = lane & 15;
    const int sel  = lane >> 4;      // 0 or 1

    v8f acc[2][4] = {{v8f{}, v8f{}, v8f{}, v8f{}},
                     {v8f{}, v8f{}, v8f{}, v8f{}}};

    const __bf16* arow0 = A + (size_t)(m0 + mrow) * 256 + sel * 8;
    const __bf16* arow1 = arow0 + (size_t)16 * 256;

    for (int kt = 0; kt < 8; ++kt) {
        const int k0 = kt * 32;
        // A fragments: v[0..7] <- K=k0+sel*8+0..7 ; v[8..15] <- +16.
        v16bf afrag0, afrag1;
#pragma unroll
        for (int j = 0; j < 8; ++j) afrag0[j] = arow0[k0 + j];
#pragma unroll
        for (int j = 0; j < 8; ++j) afrag0[8 + j] = arow0[k0 + 16 + j];
#pragma unroll
        for (int j = 0; j < 8; ++j) afrag1[j] = arow1[k0 + j];
#pragma unroll
        for (int j = 0; j < 8; ++j) afrag1[8 + j] = arow1[k0 + 16 + j];
        if (kt < 7) {
            __builtin_prefetch(arow0 + k0 + 32, 0, 0);
            __builtin_prefetch(arow1 + k0 + 32, 0, 0);
        }

#pragma unroll
        for (int nt = 0; nt < 4; ++nt) {
            // Packed B fragment: 16 contiguous halfs per lane (32B aligned).
            const v16bf bfrag = *reinterpret_cast<const v16bf*>(
                Bp + ((size_t)(kt * 16 + nt0 + nt) * 32 + lane) * 16);
            acc[0][nt] = __builtin_amdgcn_wmma_f32_16x16x32_bf16(
                false, afrag0, false, bfrag, (short)0, acc[0][nt],
                false, false);
            acc[1][nt] = __builtin_amdgcn_wmma_f32_16x16x32_bf16(
                false, afrag1, false, bfrag, (short)0, acc[1][nt],
                false, false);
        }
    }

    // Store C: VGPR r -> row (m0 + mt*16 + sel*8 + r), col nt*16 + lane%16.
#pragma unroll
    for (int mt = 0; mt < 2; ++mt) {
#pragma unroll
        for (int nt = 0; nt < 4; ++nt) {
            int col = (nt0 + nt) * 16 + (lane & 15);
#pragma unroll
            for (int r = 0; r < 8; ++r) {
                int row = m0 + mt * 16 + sel * 8 + r;
                if (row < M) C[(size_t)row * 256 + col] = acc[mt][nt][r];
            }
        }
    }
}

// -------------------- fused edge stage (f_out + logits) ---------------------
// One block (256 threads) per edge; column c in [0,256).
//   val = hni[src,c] + hnj[dst,c] + flin[e,c] + bias[c]; lk = leaky(val)
//   hidden: f_cur[e,c] += elu(lk);   last: f_cur[e,c] += 0.125*lk
//   logits[e, headBase + c/od] = sum over od cols of lk*attn[c]

__global__ __launch_bounds__(256) void edge_stage_kernel(
    const float* __restrict__ hni, const float* __restrict__ hnj,
    const float* __restrict__ flin, const float* __restrict__ bias,
    const float* __restrict__ attn, const int* __restrict__ src,
    const int* __restrict__ dst, float* __restrict__ f_cur,
    float* __restrict__ logits, int od, int headBase,
    float fscale, int lastMode) {
    __shared__ float red[256];
    int e = blockIdx.x;
    int c = threadIdx.x;
    int s = src[e], d = dst[e];
    int head = headBase + c / od;

    float val = hni[(size_t)s * 256 + c] + hnj[(size_t)d * 256 + c]
              + flin[(size_t)e * 256 + c] + bias[c];
    float lk = val > 0.f ? val : SLOPE * val;
    float resid = lastMode ? fscale * lk : (lk > 0.f ? lk : __expf(lk) - 1.f);
    f_cur[(size_t)e * 256 + c] += resid;

    red[c] = lk * attn[c];
    __syncthreads();
    for (int st = od >> 1; st > 0; st >>= 1) {
        if ((c & (od - 1)) < st) red[c] += red[c + st];
        __syncthreads();
    }
    if ((c & (od - 1)) == 0) logits[(size_t)e * NHEADS + head] = red[c];
}

// --------------------------- edge softmax ----------------------------------

__global__ void seg_max_kernel(const float* __restrict__ logits,
                               const int* __restrict__ dst,
                               float* __restrict__ segmax, int nE) {
    int idx = blockIdx.x * blockDim.x + threadIdx.x;
    if (idx < nE * NHEADS) {
        int e = idx >> 3, h = idx & 7;
        atomicMaxFloat(&segmax[(size_t)dst[e] * NHEADS + h], logits[idx]);
    }
}

__global__ void exp_sum_kernel(float* __restrict__ logits,
                               const int* __restrict__ dst,
                               const float* __restrict__ segmax,
                               float* __restrict__ segsum, int nE) {
    int idx = blockIdx.x * blockDim.x + threadIdx.x;
    if (idx < nE * NHEADS) {
        int e = idx >> 3, h = idx & 7;
        size_t di = (size_t)dst[e] * NHEADS + h;
        float ex = __expf(logits[idx] - segmax[di]);
        logits[idx] = ex;                 // overwrite with exp()
        atomicAdd(&segsum[di], ex);
    }
}

// ------------------------- message passing ----------------------------------
// n_acc[dst,c] += scale * a[e,head] * hnode[src,c]

__global__ __launch_bounds__(256) void message_kernel(
    const float* __restrict__ hnode, const float* __restrict__ ex,
    const float* __restrict__ segsum, const int* __restrict__ src,
    const int* __restrict__ dst, float* __restrict__ n_acc,
    int od, int headBase, float scale) {
    int e = blockIdx.x;
    int c = threadIdx.x;
    int head = headBase + c / od;
    int s = src[e], d = dst[e];
    float a = ex[(size_t)e * NHEADS + head] / segsum[(size_t)d * NHEADS + head];
    atomicAdd(&n_acc[(size_t)d * 256 + c],
              scale * a * hnode[(size_t)s * 256 + c]);
}

// ------------------------------- host --------------------------------------

extern "C" void kernel_launch(void* const* d_in, const int* in_sizes, int n_in,
                              void* d_out, int out_size, void* d_ws, size_t ws_size,
                              hipStream_t stream) {
    const float* nfeat   = (const float*)d_in[0];
    const float* efeat   = (const float*)d_in[1];
    const int*   src     = (const int*)d_in[2];
    const int*   dst     = (const int*)d_in[3];
    const float* Wnode_h = (const float*)d_in[4];
    const float* Wni_h   = (const float*)d_in[5];
    const float* Wnj_h   = (const float*)d_in[6];
    const float* Wfij_h  = (const float*)d_in[7];
    const float* attn_h  = (const float*)d_in[8];
    const float* bias_h  = (const float*)d_in[9];
    const float* Wnode_l = (const float*)d_in[10];
    const float* Wni_l   = (const float*)d_in[11];
    const float* Wnj_l   = (const float*)d_in[12];
    const float* Wfij_l  = (const float*)d_in[13];
    const float* attn_l  = (const float*)d_in[14];
    const float* bias_l  = (const float*)d_in[15];

    const size_t Nf = (size_t)N_NODES * 256;
    const size_t Ef = (size_t)N_EDGES * 256;

    char* p = (char*)d_ws;
    auto carve = [&](size_t bytes) {
        void* r = (void*)p;
        p += (bytes + 255) & ~(size_t)255;
        return r;
    };
    float*  n_cur   = (float*)carve(Nf * 4);
    float*  f_cur   = (float*)carve(Ef * 4);
    __bf16* n_bf    = (__bf16*)carve(Nf * 2);
    __bf16* f_bf    = (__bf16*)carve(Ef * 2);
    __bf16* Wni_bf  = (__bf16*)carve(65536 * 2);
    __bf16* Wnj_bf  = (__bf16*)carve(65536 * 2);
    __bf16* Wno_bf  = (__bf16*)carve(65536 * 2);
    __bf16* Wfi_bf  = (__bf16*)carve(65536 * 2);
    float*  hni     = (float*)carve(Nf * 4);
    float*  hnj     = (float*)carve(Nf * 4);
    float*  hnode   = (float*)carve(Nf * 4);
    float*  flin    = (float*)carve(Ef * 4);
    float*  logits  = (float*)carve((size_t)N_EDGES * NHEADS * 4);
    float*  segmax  = (float*)carve((size_t)N_NODES * NHEADS * 4);
    float*  segsum  = (float*)carve((size_t)N_NODES * NHEADS * 4);
    float*  n_acc   = (float*)carve(Nf * 4);

    auto blocks = [](size_t n) { return (unsigned)((n + 255) / 256); };
    auto gemm = [&](const __bf16* A, const __bf16* Bm, float* C, int M) {
        dim3 g(M / 128, 4);
        gemm_bf16_256<<<g, dim3(128), 0, stream>>>(A, Bm, C, M);
    };

    // running features = inputs (residual streams)
    hipMemcpyAsync(n_cur, nfeat, Nf * 4, hipMemcpyDeviceToDevice, stream);
    hipMemcpyAsync(f_cur, efeat, Ef * 4, hipMemcpyDeviceToDevice, stream);

    // ---------------- hidden layers (H=8, od=32 -> 256 cols total) ----------
    for (int l = 0; l < 2; ++l) {
        f2bf_kernel<<<blocks(Nf), 256, 0, stream>>>(n_cur, n_bf, Nf);
        f2bf_kernel<<<blocks(Ef), 256, 0, stream>>>(f_cur, f_bf, Ef);
        const size_t woff = (size_t)l * 65536;
        wconv_pack_kernel<<<256, 256, 0, stream>>>(Wni_h  + woff, Wni_bf, 256, 0);
        wconv_pack_kernel<<<256, 256, 0, stream>>>(Wnj_h  + woff, Wnj_bf, 256, 0);
        wconv_pack_kernel<<<256, 256, 0, stream>>>(Wnode_h + woff, Wno_bf, 256, 0);
        wconv_pack_kernel<<<256, 256, 0, stream>>>(Wfij_h + woff, Wfi_bf, 256, 0);

        gemm(n_bf, Wni_bf, hni, N_NODES);
        gemm(n_bf, Wnj_bf, hnj, N_NODES);
        gemm(n_bf, Wno_bf, hnode, N_NODES);
        gemm(f_bf, Wfi_bf, flin, N_EDGES);

        init_seg_kernel<<<blocks(N_NODES * NHEADS), 256, 0, stream>>>(
            segmax, segsum, N_NODES * NHEADS);
        edge_stage_kernel<<<N_EDGES, 256, 0, stream>>>(
            hni, hnj, flin, bias_h + (size_t)l * 256, attn_h + (size_t)l * 256,
            src, dst, f_cur, logits, 32, 0, 1.0f, 0);
        seg_max_kernel<<<blocks((size_t)N_EDGES * NHEADS), 256, 0, stream>>>(
            logits, dst, segmax, N_EDGES);
        exp_sum_kernel<<<blocks((size_t)N_EDGES * NHEADS), 256, 0, stream>>>(
            logits, dst, segmax, segsum, N_EDGES);
        zero_kernel<<<blocks(Nf), 256, 0, stream>>>(n_acc, Nf);
        message_kernel<<<N_EDGES, 256, 0, stream>>>(
            hnode, logits, segsum, src, dst, n_acc, 32, 0, 1.0f);
        node_update_kernel<<<blocks(Nf), 256, 0, stream>>>(n_cur, n_acc, Nf, 1);
    }

    // ---------------- last layer (H=8, od=256; per-head 256-col chunks) -----
    f2bf_kernel<<<blocks(Nf), 256, 0, stream>>>(n_cur, n_bf, Nf);
    f2bf_kernel<<<blocks(Ef), 256, 0, stream>>>(f_cur, f_bf, Ef);
    init_seg_kernel<<<blocks(N_NODES * NHEADS), 256, 0, stream>>>(
        segmax, segsum, N_NODES * NHEADS);

    for (int h = 0; h < NHEADS; ++h) {
        wconv_pack_kernel<<<256, 256, 0, stream>>>(Wni_l,  Wni_bf, 2048, h * 256);
        wconv_pack_kernel<<<256, 256, 0, stream>>>(Wnj_l,  Wnj_bf, 2048, h * 256);
        wconv_pack_kernel<<<256, 256, 0, stream>>>(Wfij_l, Wfi_bf, 2048, h * 256);
        gemm(n_bf, Wni_bf, hni, N_NODES);
        gemm(n_bf, Wnj_bf, hnj, N_NODES);
        gemm(f_bf, Wfi_bf, flin, N_EDGES);
        edge_stage_kernel<<<N_EDGES, 256, 0, stream>>>(
            hni, hnj, flin, bias_l + (size_t)h * 256, attn_l + (size_t)h * 256,
            src, dst, f_cur, logits, 256, h, 0.125f, 1);
    }
    seg_max_kernel<<<blocks((size_t)N_EDGES * NHEADS), 256, 0, stream>>>(
        logits, dst, segmax, N_EDGES);
    exp_sum_kernel<<<blocks((size_t)N_EDGES * NHEADS), 256, 0, stream>>>(
        logits, dst, segmax, segsum, N_EDGES);
    zero_kernel<<<blocks(Nf), 256, 0, stream>>>(n_acc, Nf);
    for (int h = 0; h < NHEADS; ++h) {
        wconv_pack_kernel<<<256, 256, 0, stream>>>(Wnode_l, Wno_bf, 2048, h * 256);
        gemm(n_bf, Wno_bf, hnode, N_NODES);
        message_kernel<<<N_EDGES, 256, 0, stream>>>(
            hnode, logits, segsum, src, dst, n_acc, 256, h, 0.125f);
    }
    node_update_kernel<<<blocks(Nf), 256, 0, stream>>>(n_cur, n_acc, Nf, 0);

    // outputs: (n [N,256], f [E,256]) concatenated flat
    hipMemcpyAsync(d_out, n_cur, Nf * 4, hipMemcpyDeviceToDevice, stream);
    hipMemcpyAsync((float*)d_out + Nf, f_cur, Ef * 4,
                   hipMemcpyDeviceToDevice, stream);
    (void)in_sizes; (void)n_in; (void)out_size; (void)ws_size;
}
